// MoDInfiniTransformer_40982577938962
// MI455X (gfx1250) — matrix operand order repaired and checked by reading, not compiled, not run on hardware
//
#include <hip/hip_runtime.h>
#include <hip/hip_bf16.h>

typedef __bf16 bf16;
typedef __attribute__((ext_vector_type(16))) __bf16 v16bf;
typedef __attribute__((ext_vector_type(8)))  float  v8f;

union ABFrag { v16bf v; unsigned int u[8]; };

constexpr int cB = 4, cS = 8192, cD = 1024, cH = 8, cHD = 64, cDH = 4096;
constexpr int cFULL = 2048, cSEG = 256, cNF = 4, cT = 1024;  // cT = cNF*cSEG selected/batch
constexpr int cBT = cB * cT;                                 // 4096 selected tokens
constexpr int cHK = cH * cHD;                                // 512

// ---------------------------------------------------------------------------
// CDNA5 async memory->LDS copy (ASYNCcnt-tracked DMA, bypasses VGPRs).
// Each enabled lane copies 16 bytes: LDS[vdst] = MEM[vaddr].
// ---------------------------------------------------------------------------
static __device__ __forceinline__ void async_cp16(unsigned lds_off, const void* g) {
  asm volatile("global_load_async_to_lds_b128 %0, %1, off"
               :: "v"(lds_off), "v"((unsigned long long)(size_t)g) : "memory");
}
static __device__ __forceinline__ void wait_async4() {
  asm volatile("s_wait_asynccnt 0x4" ::: "memory");
}
static __device__ __forceinline__ void wait_async0() {
  asm volatile("s_wait_asynccnt 0x0" ::: "memory");
}
static __device__ __forceinline__ unsigned ldsoff(const void* p) {
  return (unsigned)(unsigned long long)(size_t)p;   // addr[31:0] == LDS byte offset
}

// ---------------------------------------------------------------------------
// bf16 WMMA GEMM: D = alpha * A @ B^T (+ Cin) (+ bias) (ReLU?)
//   A: M x K row-major (lda).  B: N x K row-major (ldb)  -> D is M x N.
//   Batched via blockIdx.z decomposed (z/hdiv, z%hdiv).
// Block tile 128x128x32, 256 threads = 8 waves, wave tile 64x32 (4x2 WMMAs).
// Tiles staged with double-buffered GLOBAL_LOAD_ASYNC_TO_LDS_B128:
// per K-step each wave issues 2 async B128 for A + 2 for B (4 outstanding),
// pipelined with s_wait_asynccnt 4. OOB rows/cols are clamped (in-bounds
// garbage only feeds never-stored outputs; epilogue re-checks bounds).
// ---------------------------------------------------------------------------
template<bool OUTBF16, bool RELU, bool HASC, bool HASBIAS>
__global__ __launch_bounds__(256) void gemm_wmma(
    const bf16* __restrict__ A, int lda, long long aSb, long long aSh,
    const bf16* __restrict__ Bm, int ldb, long long bSb, long long bSh,
    const float* __restrict__ Cin, int ldc, long long cSb, long long cSh,
    const float* __restrict__ bias,
    void* __restrict__ Dp, int ldd, long long dSb, long long dSh,
    int M, int N, int K, float alpha, int hdiv)
{
  __shared__ __align__(16) bf16 As[2][128][32];
  __shared__ __align__(16) bf16 Bs[2][128][32];   // [n][k]

  const int tid = threadIdx.x;
  const int zb = blockIdx.z / hdiv, zh = blockIdx.z % hdiv;
  A  += zb * aSb + zh * aSh;
  Bm += zb * bSb + zh * bSh;
  const long long cOff = HASC ? (zb * cSb + zh * cSh) : 0;
  const long long dOff = zb * dSb + zh * dSh;

  const int blockM = blockIdx.y * 128;
  const int blockN = blockIdx.x * 128;
  const int wid = tid >> 5, lane = tid & 31;
  const int wm = wid & 1, wn = wid >> 1;
  const int hl = lane >> 4, l16 = lane & 15;

  // async staging geometry: 16B chunk c -> row = c>>2, col = (c&3)*8 elements.
  // wave w handles chunks w*64+lane and w*64+32+lane for each of A and B.
  const int ca0 = wid * 64 + lane, ca1 = ca0 + 32;
  const int rA0 = ca0 >> 2, cA0 = (ca0 & 3) * 8;
  const int rA1 = ca1 >> 2, cA1 = (ca1 & 3) * 8;
  const int gA0 = (blockM + rA0 < M) ? (blockM + rA0) : (M - 1);
  const int gA1 = (blockM + rA1 < M) ? (blockM + rA1) : (M - 1);
  const int gB0 = (blockN + rA0 < N) ? (blockN + rA0) : (N - 1);
  const int gB1 = (blockN + rA1 < N) ? (blockN + rA1) : (N - 1);
  const bf16* aP0 = A + (long long)gA0 * lda + cA0;
  const bf16* aP1 = A + (long long)gA1 * lda + cA1;
  const bf16* bP0 = Bm + (long long)gB0 * ldb + cA0;
  const bf16* bP1 = Bm + (long long)gB1 * ldb + cA1;
  unsigned aL[2], aH[2], bL[2], bH[2];
#pragma unroll
  for (int bi = 0; bi < 2; ++bi) {
    aL[bi] = ldsoff(&As[bi][rA0][cA0]);
    aH[bi] = ldsoff(&As[bi][rA1][cA1]);
    bL[bi] = ldsoff(&Bs[bi][rA0][cA0]);
    bH[bi] = ldsoff(&Bs[bi][rA1][cA1]);
  }
  auto issue = [&](int buf, int k0) {
    async_cp16(aL[buf], aP0 + k0);
    async_cp16(aH[buf], aP1 + k0);
    async_cp16(bL[buf], bP0 + k0);
    async_cp16(bH[buf], bP1 + k0);
  };

  v8f acc[4][2];
#pragma unroll
  for (int i = 0; i < 4; ++i)
#pragma unroll
    for (int j = 0; j < 2; ++j)
#pragma unroll
      for (int r = 0; r < 8; ++r) acc[i][j][r] = 0.0f;

  const int nk = K >> 5;
  issue(0, 0);
  for (int it = 0; it < nk; ++it) {
    const int buf = it & 1;
    if (it + 1 < nk) { issue(buf ^ 1, (it + 1) << 5); wait_async4(); }
    else             { wait_async0(); }
    __syncthreads();   // tile `buf` fully resident for all waves

    // A-frag (16-bit A 16x32): lanes 0-15 -> K 0..7 & 16..23,
    // lanes 16-31 -> K 8..15 & 24..31, two bf16 per VGPR.
    ABFrag af[4], bfr[2];
#pragma unroll
    for (int ms = 0; ms < 4; ++ms) {
      const unsigned int* ap = reinterpret_cast<const unsigned int*>(
          &As[buf][wm * 64 + ms * 16 + l16][0]);
      int kb = hl * 4;
#pragma unroll
      for (int j = 0; j < 4; ++j) { af[ms].u[j] = ap[kb + j]; af[ms].u[4 + j] = ap[8 + kb + j]; }
    }
    // B-frag: lanes 0-15 K=0..7 & 16..23, lanes 16-31 K=8..15 & 24..31, N = lane%16.
#pragma unroll
    for (int ns = 0; ns < 2; ++ns) {
      const unsigned int* bp = reinterpret_cast<const unsigned int*>(
          &Bs[buf][wn * 32 + ns * 16 + l16][0]);
      int kb = hl * 4;
#pragma unroll
      for (int j = 0; j < 4; ++j) { bfr[ns].u[j] = bp[kb + j]; bfr[ns].u[4 + j] = bp[8 + kb + j]; }
    }
#pragma unroll
    for (int ms = 0; ms < 4; ++ms)
#pragma unroll
      for (int ns = 0; ns < 2; ++ns)
        acc[ms][ns] = __builtin_amdgcn_wmma_f32_16x16x32_bf16(
            false, af[ms].v, false, bfr[ns].v, (short)0, acc[ms][ns], false, false);
    __syncthreads();   // done with `buf` before it is refilled next+1 iter
  }

  // Epilogue: C/D layout — VGPR r: lanes 0-15 M=r, lanes 16-31 M=r+8; N=lane%16.
#pragma unroll
  for (int ms = 0; ms < 4; ++ms)
#pragma unroll
    for (int ns = 0; ns < 2; ++ns) {
      int gc = blockN + wn * 32 + ns * 16 + l16;
      if (gc >= N) continue;
#pragma unroll
      for (int r = 0; r < 8; ++r) {
        int gr = blockM + wm * 64 + ms * 16 + hl * 8 + r;
        if (gr >= M) continue;
        float v = acc[ms][ns][r] * alpha;
        if (HASC) v += Cin[cOff + (long long)gr * ldc + gc];
        if (HASBIAS) v += bias[gc];
        if (RELU) v = fmaxf(v, 0.0f);
        if (OUTBF16) reinterpret_cast<bf16*>(Dp)[dOff + (long long)gr * ldd + gc] = (bf16)v;
        else         reinterpret_cast<float*>(Dp)[dOff + (long long)gr * ldd + gc] = v;
      }
    }
}

// ---------------------------------------------------------------------------
// Elementwise / helper kernels
// ---------------------------------------------------------------------------
// transpose + convert: src R x C fp32 (row-major) -> dst C x R bf16
__global__ void cvt_transpose_kernel(const float* __restrict__ s, bf16* __restrict__ d,
                                     int R, int C) {
  long long n = (long long)R * C;
  long long i = (long long)blockIdx.x * blockDim.x + threadIdx.x;
  long long st = (long long)gridDim.x * blockDim.x;
  for (; i < n; i += st) {
    int r = (int)(i / C), c = (int)(i % C);
    d[(long long)c * R + r] = (bf16)s[i];
  }
}
__global__ void fill_f32_kernel(float* d, float v, long long n) {
  long long i = (long long)blockIdx.x * blockDim.x + threadIdx.x;
  long long st = (long long)gridDim.x * blockDim.x;
  for (; i < n; i += st) d[i] = v;
}
__global__ void fill_i32_kernel(int* d, int v, long long n) {
  long long i = (long long)blockIdx.x * blockDim.x + threadIdx.x;
  long long st = (long long)gridDim.x * blockDim.x;
  for (; i < n; i += st) d[i] = v;
}

// scores = x @ w_sample + b : one wave per token (memory-bound, streams x once)
__global__ __launch_bounds__(256) void sample_scores_kernel(
    const float* __restrict__ x, const float* __restrict__ w,
    const float* __restrict__ bsc, float* __restrict__ sc_ws, float* __restrict__ sc_out) {
  __shared__ float ws[cD];
  int tid = threadIdx.x;
  for (int j = tid; j < cD; j += 256) ws[j] = w[j];
  __syncthreads();
  int wid = tid >> 5, lane = tid & 31;
  long long token = (long long)blockIdx.x * 8 + wid;
  const float* xr = x + token * cD;
  float s = 0.f;
  for (int j = 0; j < cD / 32; ++j) s += xr[lane + 32 * j] * ws[lane + 32 * j];
  for (int o = 16; o; o >>= 1) s += __shfl_xor(s, o, 32);
  if (lane == 0) { float v = s + bsc[0]; sc_ws[token] = v; sc_out[token] = v; }
}

// top-256 of 2048 per (b,f): bitonic sort desc by (val, idx-asc tiebreak),
// then bitonic index-sort of winners ascending (== jax top_k + sort).
__global__ __launch_bounds__(1024) void topk_kernel(
    const float* __restrict__ sc, int* __restrict__ sel,
    int* __restrict__ selpos, float* __restrict__ mask) {
  __shared__ float v[cFULL];
  __shared__ int  ix[cFULL];
  int tid = threadIdx.x;
  int b = blockIdx.x / cNF, f = blockIdx.x % cNF;
  const float* s = sc + (long long)b * cS + f * cFULL;
  for (int i = tid; i < cFULL; i += 1024) { v[i] = s[i]; ix[i] = i; }
  for (int k = 2; k <= cFULL; k <<= 1)
    for (int j = k >> 1; j > 0; j >>= 1) {
      __syncthreads();
      for (int i = tid; i < cFULL; i += 1024) {
        int p = i ^ j;
        if (p > i) {
          bool up = ((i & k) == 0);
          bool g = (v[i] > v[p]) || (v[i] == v[p] && ix[i] < ix[p]);
          if (up ? !g : g) {
            float tv = v[i]; v[i] = v[p]; v[p] = tv;
            int ti = ix[i]; ix[i] = ix[p]; ix[p] = ti;
          }
        }
      }
    }
  for (int k = 2; k <= cSEG; k <<= 1)
    for (int j = k >> 1; j > 0; j >>= 1) {
      __syncthreads();
      if (tid < 128)
        for (int i = tid; i < cSEG; i += 128) {
          int p = i ^ j;
          if (p > i) {
            bool up = ((i & k) == 0);
            bool g = ix[i] > ix[p];
            if (up ? g : !g) { int t = ix[i]; ix[i] = ix[p]; ix[p] = t; }
          }
        }
    }
  __syncthreads();
  if (tid < cSEG) {
    int tok = f * cFULL + ix[tid];
    int jpos = f * cSEG + tid;
    sel[b * cT + jpos] = tok;
    selpos[(long long)b * cS + tok] = jpos;
    mask[(long long)b * cS + tok] = 1.0f;
  }
}

__global__ __launch_bounds__(256) void gather_kernel(
    const float* __restrict__ x, const int* __restrict__ sel, bf16* __restrict__ xsel) {
  int row = blockIdx.x; int b = row >> 10;
  int tok = sel[row];
  const float* src = x + ((long long)b * cS + tok) * cD;
  bf16* dst = xsel + (long long)row * cD;
  for (int i = threadIdx.x; i < cD; i += 256) dst[i] = (bf16)src[i];
}

static __device__ __forceinline__ float elup1(float x) { // elu(x)+1
  return x > 0.f ? x + 1.f : __expf(x);
}

// sq = elu(q)+1 (token-major); skT, vT transposed per (b,h): [64][cT]
__global__ __launch_bounds__(256) void prep_qkv_kernel(
    const bf16* __restrict__ qb, const bf16* __restrict__ kb, const bf16* __restrict__ vb,
    bf16* __restrict__ sqb, bf16* __restrict__ sktb, bf16* __restrict__ vtb) {
  long long i = (long long)blockIdx.x * 256 + threadIdx.x;   // over cBT*cHK
  int c = (int)(i % cHK); long long tg = i / cHK;
  int h = c >> 6, d = c & 63;
  int b = (int)(tg >> 10); int tl = (int)(tg & 1023);
  sqb[i] = (bf16)elup1((float)qb[i]);
  long long tofs = (((long long)(b * cH + h) * cHD + d) * cT) + tl;
  sktb[tofs] = (bf16)elup1((float)kb[i]);
  vtb[tofs]  = vb[i];
}

// softmax over 256-wide rows, one wave per row, fp32 in -> bf16 P out
__global__ __launch_bounds__(256) void softmax_kernel(
    const float* __restrict__ S, bf16* __restrict__ P) {
  int wid = threadIdx.x >> 5, lane = threadIdx.x & 31;
  long long row = (long long)blockIdx.x * 8 + wid;
  const float* sr = S + row * cSEG;
  float vals[8]; float mx = -3.4e38f;
#pragma unroll
  for (int j = 0; j < 8; ++j) { vals[j] = sr[lane + 32 * j]; mx = fmaxf(mx, vals[j]); }
  for (int o = 16; o; o >>= 1) mx = fmaxf(mx, __shfl_xor(mx, o, 32));
  float sum = 0.f;
#pragma unroll
  for (int j = 0; j < 8; ++j) { vals[j] = __expf(vals[j] - mx); sum += vals[j]; }
  for (int o = 16; o; o >>= 1) sum += __shfl_xor(sum, o, 32);
  float inv = 1.0f / sum;
  bf16* pr = P + row * cSEG;
#pragma unroll
  for (int j = 0; j < 8; ++j) pr[lane + 32 * j] = (bf16)(vals[j] * inv);
}

// att = gate * (num/denom) + (1-gate) * att_dot ; denom = sq . z (old z)
__global__ __launch_bounds__(256) void combine_kernel(
    const float* __restrict__ num, const float* __restrict__ adot,
    const bf16* __restrict__ sqb, const float* __restrict__ zv,
    const float* __restrict__ betas, bf16* __restrict__ att, int seg) {
  int wid = threadIdx.x >> 5, lane = threadIdx.x & 31;
  int wg = blockIdx.x * 8 + wid;
  int z = wg >> 8, s = wg & 255;
  int b = z >> 3, h = z & 7;
  const bf16* sqr = sqb + ((long long)(b * cT + seg * cSEG + s) * cHK) + h * cHD;
  const float* zr = zv + z * cHD;
  float d = (float)sqr[2 * lane] * zr[2 * lane] + (float)sqr[2 * lane + 1] * zr[2 * lane + 1];
  for (int o = 16; o; o >>= 1) d += __shfl_xor(d, o, 32);
  float invd = 1.0f / d;
  long long nbase = ((long long)z * cSEG + s) * cHD;
  bf16* ar = att + ((long long)(b * cT + seg * cSEG + s) * cHK) + h * cHD;
#pragma unroll
  for (int j = 0; j < 2; ++j) {
    int vv = lane + 32 * j;
    float g = 1.0f / (1.0f + __expf(-betas[h * cHD + vv]));
    float o = g * num[nbase + vv] * invd + (1.0f - g) * adot[nbase + vv];
    ar[vv] = (bf16)o;
  }
}

// z += sum_t sk ; memT(bf16,[v][k]) = mem(fp32,[k][v])  (post-update refresh)
__global__ __launch_bounds__(256) void zmem_kernel(
    const bf16* __restrict__ sktb, const float* __restrict__ memf,
    float* __restrict__ zv, bf16* __restrict__ memT, int seg) {
  int z = blockIdx.x, tid = threadIdx.x;
  if (tid < cHD) {
    const bf16* p = sktb + ((long long)(z * cHD + tid) * cT) + seg * cSEG;
    float s = 0.f;
    for (int t = 0; t < cSEG; ++t) s += (float)p[t];
    zv[z * cHD + tid] += s;
  }
  for (int i = tid; i < cHD * cHD; i += 256) {
    int kd = i >> 6, vx = i & 63;
    memT[(long long)z * 4096 + vx * 64 + kd] = (bf16)memf[(long long)z * 4096 + kd * 64 + vx];
  }
}

// fused scatter-add residual + LayerNorm, one block per token (streams x -> y)
__global__ __launch_bounds__(256) void scatter_ln_kernel(
    const float* __restrict__ x, const float* __restrict__ h2,
    const int* __restrict__ selpos, const float* __restrict__ gamma,
    const float* __restrict__ beta, float* __restrict__ y) {
  __shared__ float r1[8], r2[8], mv[2];
  long long row = blockIdx.x;
  int b = (int)(row >> 13);
  int sp = selpos[row];
  long long xbase = row * cD;
  long long hbase = (sp >= 0) ? ((long long)(b * cT + sp) * cD) : 0;
  int tid = threadIdx.x;
  float vals[4]; float s = 0.f, s2 = 0.f;
#pragma unroll
  for (int j = 0; j < 4; ++j) {
    int c = tid + j * 256;
    float v = x[xbase + c];
    if (sp >= 0) v += h2[hbase + c];
    vals[j] = v; s += v; s2 += v * v;
  }
  int wid = tid >> 5, lane = tid & 31;
  for (int o = 16; o; o >>= 1) { s += __shfl_xor(s, o, 32); s2 += __shfl_xor(s2, o, 32); }
  if (lane == 0) { r1[wid] = s; r2[wid] = s2; }
  __syncthreads();
  if (tid == 0) {
    float ts = 0.f, ts2 = 0.f;
    for (int i = 0; i < 8; ++i) { ts += r1[i]; ts2 += r2[i]; }
    float mu = ts / cD;
    float var = ts2 / cD - mu * mu;
    mv[0] = mu; mv[1] = rsqrtf(var + 1e-5f);
  }
  __syncthreads();
  float mu = mv[0], rstd = mv[1];
#pragma unroll
  for (int j = 0; j < 4; ++j) {
    int c = tid + j * 256;
    y[xbase + c] = (vals[j] - mu) * rstd * gamma[c] + beta[c];
  }
}

// ---------------------------------------------------------------------------
extern "C" void kernel_launch(void* const* d_in, const int* in_sizes, int n_in,
                              void* d_out, int out_size, void* d_ws, size_t ws_size,
                              hipStream_t stream) {
  (void)in_sizes; (void)n_in; (void)out_size; (void)ws_size;
  const float* x        = (const float*)d_in[0];
  const float* w_sample = (const float*)d_in[1];
  const float* b_sample = (const float*)d_in[2];
  const float* wq       = (const float*)d_in[3];
  const float* wk       = (const float*)d_in[4];
  const float* wv       = (const float*)d_in[5];
  const float* w_out    = (const float*)d_in[6];
  const float* betas    = (const float*)d_in[7];
  const float* w1       = (const float*)d_in[8];
  const float* b1       = (const float*)d_in[9];
  const float* w2       = (const float*)d_in[10];
  const float* b2       = (const float*)d_in[11];
  const float* ln_g     = (const float*)d_in[12];
  const float* ln_b     = (const float*)d_in[13];

  float* y_out    = (float*)d_out;
  float* mask_out = y_out + (long long)cB * cS * cD;
  float* sc_out   = mask_out + (long long)cB * cS;

  char* ws = (char*)d_ws;
  size_t off = 0;
  auto alloc = [&](size_t bytes) {
    off = (off + 255) & ~(size_t)255; size_t c = off; off += bytes; return c;
  };
  float* sc_ws  = (float*)(ws + alloc(4ll * cB * cS));
  int*   sel    = (int*)  (ws + alloc(4ll * cB * cT));
  int*   selpos = (int*)  (ws + alloc(4ll * cB * cS));
  bf16*  wqTb   = (bf16*) (ws + alloc(2ll * cD * cHK));   // 512 x 1024
  bf16*  wkTb   = (bf16*) (ws + alloc(2ll * cD * cHK));
  bf16*  wvTb   = (bf16*) (ws + alloc(2ll * cD * cHK));
  bf16*  woutTb = (bf16*) (ws + alloc(2ll * cHK * cD));   // 1024 x 512
  bf16*  w1Tb   = (bf16*) (ws + alloc(2ll * cD * cDH));   // 4096 x 1024
  bf16*  w2Tb   = (bf16*) (ws + alloc(2ll * cDH * cD));   // 1024 x 4096
  bf16*  xsel   = (bf16*) (ws + alloc(2ll * cBT * cD));
  bf16*  qb     = (bf16*) (ws + alloc(2ll * cBT * cHK));
  bf16*  kbb    = (bf16*) (ws + alloc(2ll * cBT * cHK));
  bf16*  vbb    = (bf16*) (ws + alloc(2ll * cBT * cHK));
  bf16*  sqb    = (bf16*) (ws + alloc(2ll * cBT * cHK));
  bf16*  sktb   = (bf16*) (ws + alloc(2ll * cBT * cHK));
  bf16*  vtb    = (bf16*) (ws + alloc(2ll * cBT * cHK));
  float* scoresf= (float*)(ws + alloc(4ll * 32 * cSEG * cSEG));
  bf16*  Pbuf   = (bf16*) (ws + alloc(2ll * 32 * cSEG * cSEG));
  float* numb   = (float*)(ws + alloc(4ll * 32 * cSEG * cHD));
  float* adotb  = (float*)(ws + alloc(4ll * 32 * cSEG * cHD));
  float* memf   = (float*)(ws + alloc(4ll * 32 * cHD * cHD));
  bf16*  memT   = (bf16*) (ws + alloc(2ll * 32 * cHD * cHD));
  float* zv     = (float*)(ws + alloc(4ll * 32 * cHD));
  bf16*  attb   = (bf16*) (ws + alloc(2ll * cBT * cHK));
  bf16*  abuf   = (bf16*) (ws + alloc(2ll * cBT * cD));
  bf16*  h1     = (bf16*) (ws + alloc(2ll * cBT * cDH));
  float* h2     = (float*)(ws + alloc(4ll * cBT * cD));

  // weights -> transposed (N x K) bf16, so all GEMMs are A @ B^T with
  // async-LDS-stageable row-major tiles on both operands
  cvt_transpose_kernel<<<512, 256, 0, stream>>>(wq, wqTb, cD, cHK);
  cvt_transpose_kernel<<<512, 256, 0, stream>>>(wk, wkTb, cD, cHK);
  cvt_transpose_kernel<<<512, 256, 0, stream>>>(wv, wvTb, cD, cHK);
  cvt_transpose_kernel<<<512, 256, 0, stream>>>(w_out, woutTb, cHK, cD);
  cvt_transpose_kernel<<<2048, 256, 0, stream>>>(w1, w1Tb, cD, cDH);
  cvt_transpose_kernel<<<2048, 256, 0, stream>>>(w2, w2Tb, cDH, cD);
  // state init
  fill_f32_kernel<<<256, 256, 0, stream>>>(mask_out, 0.f, (long long)cB * cS);
  fill_i32_kernel<<<256, 256, 0, stream>>>(selpos, -1, (long long)cB * cS);
  fill_f32_kernel<<<64, 256, 0, stream>>>(memf, 0.f, 32ll * cHD * cHD);
  fill_i32_kernel<<<64, 256, 0, stream>>>((int*)memT, 0, 32ll * cHD * cHD / 2);
  fill_f32_kernel<<<8, 256, 0, stream>>>(zv, 1.0f / 64.0f, 32ll * cHD);

  sample_scores_kernel<<<cB * cS / 8, 256, 0, stream>>>(x, w_sample, b_sample, sc_ws, sc_out);
  topk_kernel<<<cB * cNF, 1024, 0, stream>>>(sc_ws, sel, selpos, mask_out);
  gather_kernel<<<cBT, 256, 0, stream>>>(x, sel, xsel);

  // Q/K/V projections: (4096 x 1024) @ (512 x 1024)^T
  dim3 gQ(cHK / 128, cBT / 128, 1);
  gemm_wmma<true, false, false, false><<<gQ, 256, 0, stream>>>(
      xsel, cD, 0, 0, wqTb, cD, 0, 0, nullptr, 0, 0, 0, nullptr,
      qb, cHK, 0, 0, cBT, cHK, cD, 1.0f, 1);
  gemm_wmma<true, false, false, false><<<gQ, 256, 0, stream>>>(
      xsel, cD, 0, 0, wkTb, cD, 0, 0, nullptr, 0, 0, 0, nullptr,
      kbb, cHK, 0, 0, cBT, cHK, cD, 1.0f, 1);
  gemm_wmma<true, false, false, false><<<gQ, 256, 0, stream>>>(
      xsel, cD, 0, 0, wvTb, cD, 0, 0, nullptr, 0, 0, 0, nullptr,
      vbb, cHK, 0, 0, cBT, cHK, cD, 1.0f, 1);

  prep_qkv_kernel<<<cBT * cHK / 256, 256, 0, stream>>>(qb, kbb, vbb, sqb, sktb, vtb);

  for (int seg = 0; seg < cNF; ++seg) {
    long long segTokOff = (long long)seg * cSEG * cHK;
    // scores = Q_seg @ K_seg^T / sqrt(dk), batched over (b,h)
    gemm_wmma<false, false, false, false><<<dim3(2, 2, 32), 256, 0, stream>>>(
        qb + segTokOff, cHK, (long long)cT * cHK, cHD,
        kbb + segTokOff, cHK, (long long)cT * cHK, cHD,
        nullptr, 0, 0, 0, nullptr,
        scoresf, cSEG, (long long)cH * cSEG * cSEG, (long long)cSEG * cSEG,
        cSEG, cSEG, cHD, 0.125f, cH);
    softmax_kernel<<<32 * cSEG / 8, 256, 0, stream>>>(scoresf, Pbuf);
    // att_dot = P @ V_seg   (B operand = vT slice, N x K row-major)
    gemm_wmma<false, false, false, false><<<dim3(1, 2, 32), 256, 0, stream>>>(
        Pbuf, cSEG, (long long)cH * cSEG * cSEG, (long long)cSEG * cSEG,
        vtb + seg * cSEG, cT, (long long)cH * cHD * cT, (long long)cHD * cT,
        nullptr, 0, 0, 0, nullptr,
        adotb, cHD, (long long)cH * cSEG * cHD, (long long)cSEG * cHD,
        cSEG, cHD, cSEG, 1.0f, cH);
    // num = sq_seg @ mem_old   (B operand = memT, N x K row-major)
    gemm_wmma<false, false, false, false><<<dim3(1, 2, 32), 256, 0, stream>>>(
        sqb + segTokOff, cHK, (long long)cT * cHK, cHD,
        memT, cHD, (long long)cH * cHD * cHD, (long long)cHD * cHD,
        nullptr, 0, 0, 0, nullptr,
        numb, cHD, (long long)cH * cSEG * cHD, (long long)cSEG * cHD,
        cSEG, cHD, cHD, 1.0f, cH);
    combine_kernel<<<1024, 256, 0, stream>>>(numb, adotb, sqb, zv, betas, attb, seg);
    // mem += sk^T @ V_seg  (fp32 accumulate into memf)
    gemm_wmma<false, false, true, false><<<dim3(1, 1, 32), 256, 0, stream>>>(
        sktb + seg * cSEG, cT, (long long)cH * cHD * cT, (long long)cHD * cT,
        vtb + seg * cSEG, cT, (long long)cH * cHD * cT, (long long)cHD * cT,
        memf, cHD, (long long)cH * cHD * cHD, (long long)cHD * cHD,
        nullptr,
        memf, cHD, (long long)cH * cHD * cHD, (long long)cHD * cHD,
        cHD, cHD, cSEG, 1.0f, cH);
    zmem_kernel<<<32, 256, 0, stream>>>(sktb, memf, zv, memT, seg);
  }

  // out projection: (4096 x 512) @ (1024 x 512)^T
  gemm_wmma<true, false, false, false><<<dim3(8, 32, 1), 256, 0, stream>>>(
      attb, cHK, 0, 0, woutTb, cHK, 0, 0, nullptr, 0, 0, 0, nullptr,
      abuf, cD, 0, 0, cBT, cD, cHK, 1.0f, 1);
  // MLP1: relu(a @ w1 + b1)
  gemm_wmma<true, true, false, true><<<dim3(32, 32, 1), 256, 0, stream>>>(
      abuf, cD, 0, 0, w1Tb, cD, 0, 0, nullptr, 0, 0, 0, b1,
      h1, cDH, 0, 0, cBT, cDH, cD, 1.0f, 1);
  // MLP2: h1 @ w2 + b2 (fp32 out for residual)
  gemm_wmma<false, false, false, true><<<dim3(8, 32, 1), 256, 0, stream>>>(
      h1, cDH, 0, 0, w2Tb, cDH, 0, 0, nullptr, 0, 0, 0, b2,
      h2, cD, 0, 0, cBT, cD, cDH, 1.0f, 1);

  scatter_ln_kernel<<<cB * cS, 256, 0, stream>>>(x, h2, selpos, ln_g, ln_b, y_out);
}
